// MultiHeadAttention_57320633532631
// MI455X (gfx1250) — matrix operand order repaired and checked by reading, not compiled, not run on hardware
//
#include <hip/hip_runtime.h>

#define D_MODEL 1024
#define N_HEAD  16
#define D_HEAD  64
#define BATCH   2
#define SEQ     2048
#define ROWS    (BATCH * SEQ)   /* 4096 */

typedef __attribute__((ext_vector_type(16))) __bf16 v16bf;
typedef __attribute__((ext_vector_type(8)))  float  v8f;

#define AI __device__ __attribute__((always_inline)) static inline

// ---------- TDM availability ----------
#if __has_builtin(__builtin_amdgcn_tensor_load_to_lds) && \
    __has_builtin(__builtin_amdgcn_s_wait_tensorcnt)
#define TDM_OK 1
#endif

#ifdef TDM_OK
typedef __attribute__((ext_vector_type(4))) unsigned tdm_v4u;
typedef __attribute__((ext_vector_type(8))) int      tdm_v8i;
typedef __attribute__((ext_vector_type(4))) int      tdm_v4i;

// Issue a 2D TDM tile load: tile_d0 elements/row (2B each), tile_d1 rows,
// global row stride = stride_d0 elements. LDS destination gets 4 DWORDs of
// padding after every 16 DWORDs (=> 40-half row pitch, matches LDT below).
AI void tdm_load_tile_2d(unsigned lds_off_bytes, const void* gptr,
                         unsigned tile_d0, unsigned tile_d1,
                         unsigned tensor_d0, unsigned tensor_d1,
                         unsigned stride_d0) {
  unsigned long long ga = (unsigned long long)(size_t)gptr;
  tdm_v4u g0;
  g0[0] = 1u;                                            // count=1 (valid), user D#
  g0[1] = lds_off_bytes;                                 // lds_addr
  g0[2] = (unsigned)(ga & 0xFFFFFFFFu);                  // global_addr[31:0]
  g0[3] = (unsigned)((ga >> 32) & 0x01FFFFFFu)           // global_addr[56:32]
        | (2u << 30);                                    // type = 2 ("image")
  tdm_v8i g1;
  g1[0] = (1 << 16)                                      // data_size = 2 bytes
        | (1 << 20)                                      // pad_enable
        | (3 << 22)                                      // pad_interval: 16 DWORDs
        | (3 << 25);                                     // pad_amount: 4 DWORDs
  g1[1] = (int)((tensor_d0 & 0xFFFFu) << 16);            // tensor_dim0[15:0]
  g1[2] = (int)(((tensor_d0 >> 16) & 0xFFFFu)            // tensor_dim0[31:16]
        | ((tensor_d1 & 0xFFFFu) << 16));                // tensor_dim1[15:0]
  g1[3] = (int)(((tensor_d1 >> 16) & 0xFFFFu)            // tensor_dim1[31:16]
        | ((tile_d0 & 0xFFFFu) << 16));                  // tile_dim0
  g1[4] = (int)(tile_d1 & 0xFFFFu);                      // tile_dim1 (tile_dim2=0)
  g1[5] = (int)stride_d0;                                // tensor_dim0_stride[31:0]
  g1[6] = 0;                                             // stride0 hi / stride1 lo
  g1[7] = 0;
  tdm_v4i gz = {0, 0, 0, 0};
#if __clang_major__ >= 23
  tdm_v8i gz8 = {0, 0, 0, 0, 0, 0, 0, 0};
  __builtin_amdgcn_tensor_load_to_lds(g0, g1, gz, gz, gz8, 0);
#else
  __builtin_amdgcn_tensor_load_to_lds(g0, g1, gz, gz, 0);
#endif
}
#endif  // TDM_OK

// ---------- helpers ----------

AI unsigned short f32_to_bf16(float f) {
  unsigned u = __float_as_uint(f);
  unsigned r = 0x7FFFu + ((u >> 16) & 1u);   // round-to-nearest-even
  return (unsigned short)((u + r) >> 16);
}

// A-fragment (16x32 bf16, MxK). base points at (row0,k0); ld = row stride in halfs.
// lanes 0-15 hold M=lane, K {0..15}; lanes 16-31 hold M=lane-16,
// K {16..31}.
AI v16bf load_a_frag(const unsigned short* base, int lane, int ld) {
  int m  = lane & 15;
  int hi = (lane >> 4) & 1;
  const unsigned short* p0 = base + m * ld + hi * 16;
  v16bf r;
  ((uint4*)&r)[0] = *(const uint4*)p0;
  ((uint4*)&r)[1] = *(const uint4*)(p0 + 8);
  return r;
}

// B-fragment (32x16 bf16, KxN), column n stored K-contiguous at base + n*ld.
AI v16bf load_b_frag(const unsigned short* base, int lane, int ld) {
  int n  = lane & 15;
  int hi = (lane >> 4) & 1;
  const unsigned short* p = base + n * ld + hi * 16;
  v16bf r;
  ((uint4*)&r)[0] = *(const uint4*)p;
  ((uint4*)&r)[1] = *(const uint4*)(p + 8);
  return r;
}

AI v8f wmma_bf16(v16bf a, v16bf b, v8f c) {
  return __builtin_amdgcn_wmma_f32_16x16x32_bf16(
      false, a, false, b, (short)0, c, false, false);
}

// ---------- cast kernel ----------

__global__ __launch_bounds__(256)
void cast_f32_to_bf16_kernel(const float* __restrict__ src,
                             unsigned short* __restrict__ dst, int n) {
  int i = blockIdx.x * blockDim.x + threadIdx.x;
  int stride = gridDim.x * blockDim.x;
  for (; i < n; i += stride) dst[i] = f32_to_bf16(src[i]);
}

// ---------- GEMM: out[m,n] = sum_k A[m,k]*W[n,k] + bias[n]  (X @ W^T + b) ----------
// Block tile 128x128, 8 waves as 4(M) x 2(N), wave tile 32x64 -> 8 WMMA/k-step/wave.
// TDM path: wave 0 DMAs both tiles (double-buffered, pipelined one step ahead);
// fallback: cooperative vector staging.
// NOTE: all LDS pointers/offsets are kept as SCALAR locals (no aggregates) so no
// addrspacecast constants leak into static initializers (ld.lld limitation).

#define BM 128
#define BN 128
#define BK 32
#define LDT (BK + 8)
#define TILE_H (BM * LDT)    /* halfs per staged tile */

__global__ __launch_bounds__(256)
void gemm_xwT_bias_kernel(const unsigned short* __restrict__ A,   // [M][K] bf16
                          const unsigned short* __restrict__ W,   // [N][K] bf16
                          const float* __restrict__ bias,         // [N]
                          unsigned short* __restrict__ out_bf16,  // [M][N] or null
                          float* __restrict__ out_f32,            // [M][N] or null
                          int M, int K, int N) {
  __shared__ unsigned short smem[4 * TILE_H];   // A0 A1 B0 B1

  const int tid  = threadIdx.x;
  const int lane = tid & 31;
  const int wave = tid >> 5;
  const int wr   = wave >> 1;          // 0..3
  const int wc   = wave & 1;           // 0..1
  const int row0 = blockIdx.y * BM;
  const int col0 = blockIdx.x * BN;
  const int hi   = (lane >> 4) & 1;
  const int ln   = lane & 15;

  v8f acc[2][4] = {};
  const int nstep = K / BK;

#ifdef TDM_OK
  const bool issuer = (tid < 32);
  // Generic->LDS pointer low 32 bits are the LDS byte offset (ISA aperture rule).
  const unsigned lds0  = (unsigned)(size_t)&smem[0];
  const unsigned offA0 = lds0;
  const unsigned offA1 = lds0 + TILE_H * 2u;
  const unsigned offB0 = lds0 + 2u * TILE_H * 2u;
  const unsigned offB1 = lds0 + 3u * TILE_H * 2u;

  if (issuer) {
    tdm_load_tile_2d(offA0, A + (size_t)row0 * K, BK, BM,
                     (unsigned)K, (unsigned)M, (unsigned)K);
    tdm_load_tile_2d(offB0, W + (size_t)col0 * K, BK, BN,
                     (unsigned)K, (unsigned)N, (unsigned)K);
  }
  for (int i = 0; i < nstep; ++i) {
    if (issuer) __builtin_amdgcn_s_wait_tensorcnt(0);   // tile i landed
    __syncthreads();
    const int cur = i & 1;
    if (issuer && (i + 1) < nstep) {                    // prefetch tile i+1
      const int k0 = (i + 1) * BK;
      tdm_load_tile_2d(cur ? offA0 : offA1, A + (size_t)row0 * K + k0, BK, BM,
                       (unsigned)K, (unsigned)M, (unsigned)K);
      tdm_load_tile_2d(cur ? offB0 : offB1, W + (size_t)col0 * K + k0, BK, BN,
                       (unsigned)K, (unsigned)N, (unsigned)K);
    }
    const unsigned short* sA = smem + (cur ? TILE_H : 0);
    const unsigned short* sB = smem + 2 * TILE_H + (cur ? TILE_H : 0);
    v16bf a0 = load_a_frag(sA + (wr * 32 + 0)  * LDT, lane, LDT);
    v16bf a1 = load_a_frag(sA + (wr * 32 + 16) * LDT, lane, LDT);
#pragma unroll
    for (int j = 0; j < 4; ++j) {
      v16bf b = load_b_frag(sB + (wc * 64 + j * 16) * LDT, lane, LDT);
      acc[0][j] = wmma_bf16(a0, b, acc[0][j]);
      acc[1][j] = wmma_bf16(a1, b, acc[1][j]);
    }
  }
#else
  unsigned short* sA = smem;                 // single-buffered fallback
  unsigned short* sB = smem + 2 * TILE_H;
  for (int i = 0; i < nstep; ++i) {
    const int k0 = i * BK;
    __syncthreads();
#pragma unroll
    for (int it = 0; it < 2; ++it) {   // A tile 128x32: 512 chunks, 2/thread
      int idx = tid + it * 256;
      int r = idx >> 2;
      int c = (idx & 3) * 8;
      uint4 d0 = *(const uint4*)(A + (size_t)(row0 + r) * K + k0 + c);
      *(uint4*)(sA + r * LDT + c) = d0;
    }
#pragma unroll
    for (int it = 0; it < 2; ++it) {   // W tile 128x32
      int idx = tid + it * 256;
      int r = idx >> 2;
      int c = (idx & 3) * 8;
      uint4 d0 = *(const uint4*)(W + (size_t)(col0 + r) * K + k0 + c);
      *(uint4*)(sB + r * LDT + c) = d0;
    }
    __syncthreads();
    v16bf a0 = load_a_frag(sA + (wr * 32 + 0)  * LDT, lane, LDT);
    v16bf a1 = load_a_frag(sA + (wr * 32 + 16) * LDT, lane, LDT);
#pragma unroll
    for (int j = 0; j < 4; ++j) {
      v16bf b = load_b_frag(sB + (wc * 64 + j * 16) * LDT, lane, LDT);
      acc[0][j] = wmma_bf16(a0, b, acc[0][j]);
      acc[1][j] = wmma_bf16(a1, b, acc[1][j]);
    }
  }
#endif

#pragma unroll
  for (int i = 0; i < 2; ++i)
#pragma unroll
    for (int j = 0; j < 4; ++j) {
      int rbase = row0 + wr * 32 + i * 16 + hi * 8;
      int col   = col0 + wc * 64 + j * 16 + ln;
      float bv  = bias[col];
#pragma unroll
      for (int r = 0; r < 8; ++r) {
        float val = acc[i][j][r] + bv;
        size_t o = (size_t)(rbase + r) * N + col;
        if (out_f32) out_f32[o] = val;
        else         out_bf16[o] = f32_to_bf16(val);
      }
    }
}

// ---------- Flash attention: per (b,h), O = softmax(Q K^T / 8) V ----------

#define QT  128
#define KT  32
#define PLD (KT + 8)

__global__ __launch_bounds__(256)
void flash_attn_kernel(const unsigned short* __restrict__ Qh,
                       const unsigned short* __restrict__ Kh,
                       const unsigned short* __restrict__ Vh,
                       unsigned short* __restrict__ Oh) {
  __shared__ unsigned short sVt[D_HEAD * PLD];       // V^T: [d][key]
  __shared__ unsigned short sP[8 * 16 * PLD];        // per-wave 16x32 P tile

  const int tid  = threadIdx.x;
  const int lane = tid & 31;
  const int wave = tid >> 5;
  const int b    = blockIdx.z;
  const int h    = blockIdx.y;
  const int q0   = blockIdx.x * QT + wave * 16;
  const int hi   = (lane >> 4) & 1;
  const int ln   = lane & 15;

  const unsigned short* Qbase = Qh + ((size_t)b * SEQ) * D_MODEL + h * D_HEAD;
  const unsigned short* Kbase = Kh + ((size_t)b * SEQ) * D_MODEL + h * D_HEAD;
  const unsigned short* Vbase = Vh + ((size_t)b * SEQ) * D_MODEL + h * D_HEAD;
  unsigned short* myP = sP + wave * 16 * PLD;

  v16bf aq[2];
#pragma unroll
  for (int kk = 0; kk < 2; ++kk)
    aq[kk] = load_a_frag(Qbase + (size_t)q0 * D_MODEL + kk * 32, lane, D_MODEL);

  v8f o[4] = {};
  float mrow[8], lrow[8];
#pragma unroll
  for (int r = 0; r < 8; ++r) { mrow[r] = -1e30f; lrow[r] = 0.0f; }

  for (int kb = 0; kb < SEQ; kb += KT) {
    __syncthreads();
    {   // stage V[32 keys][64 d] transposed into sVt[d][key]
      int key = tid >> 3;
      int d0  = (tid & 7) * 8;
      uint4 vd = *(const uint4*)(Vbase + (size_t)(kb + key) * D_MODEL + d0);
      unsigned short tmp[8];
      *(uint4*)tmp = vd;
#pragma unroll
      for (int e = 0; e < 8; ++e) sVt[(d0 + e) * PLD + key] = tmp[e];
    }
    __syncthreads();

    v8f s0 = {}, s1 = {};
#pragma unroll
    for (int kk = 0; kk < 2; ++kk) {
      const unsigned short* pk0 =
          Kbase + (size_t)(kb + 0  + ln) * D_MODEL + kk * 32 + hi * 16;
      const unsigned short* pk1 =
          Kbase + (size_t)(kb + 16 + ln) * D_MODEL + kk * 32 + hi * 16;
      v16bf bk0, bk1;
      ((uint4*)&bk0)[0] = *(const uint4*)pk0;
      ((uint4*)&bk0)[1] = *(const uint4*)(pk0 + 8);
      ((uint4*)&bk1)[0] = *(const uint4*)pk1;
      ((uint4*)&bk1)[1] = *(const uint4*)(pk1 + 8);
      s0 = wmma_bf16(aq[kk], bk0, s0);
      s1 = wmma_bf16(aq[kk], bk1, s1);
    }

    float tm[8];
#pragma unroll
    for (int r = 0; r < 8; ++r) {
      s0[r] *= 0.125f;
      s1[r] *= 0.125f;
      tm[r] = fmaxf(s0[r], s1[r]);
    }
#pragma unroll
    for (int r = 0; r < 8; ++r)
#pragma unroll
      for (int msk = 1; msk < 16; msk <<= 1)
        tm[r] = fmaxf(tm[r], __shfl_xor(tm[r], msk, 32));

    float ps[8];
#pragma unroll
    for (int r = 0; r < 8; ++r) {
      float mn = fmaxf(mrow[r], tm[r]);
      float sc = __expf(mrow[r] - mn);
      mrow[r] = mn;
      float p0 = __expf(s0[r] - mn);
      float p1 = __expf(s1[r] - mn);
      s0[r] = p0; s1[r] = p1;
      ps[r] = p0 + p1;
      lrow[r] *= sc;
      o[0][r] *= sc; o[1][r] *= sc; o[2][r] *= sc; o[3][r] *= sc;
    }
#pragma unroll
    for (int r = 0; r < 8; ++r) {
#pragma unroll
      for (int msk = 1; msk < 16; msk <<= 1)
        ps[r] += __shfl_xor(ps[r], msk, 32);
      lrow[r] += ps[r];
    }

#pragma unroll
    for (int r = 0; r < 8; ++r) {
      int prow = r + 8 * hi;
      myP[prow * PLD + ln]      = f32_to_bf16(s0[r]);
      myP[prow * PLD + 16 + ln] = f32_to_bf16(s1[r]);
    }
    v16bf ap = load_a_frag(myP, lane, PLD);   // same-wave DS ops are in-order

#pragma unroll
    for (int j2 = 0; j2 < 4; ++j2) {
      v16bf bv = load_b_frag(sVt + (j2 * 16) * PLD, lane, PLD);
      o[j2] = wmma_bf16(ap, bv, o[j2]);
    }
  }

  unsigned short* Obase = Oh + ((size_t)b * SEQ + q0) * D_MODEL + h * D_HEAD;
#pragma unroll
  for (int j2 = 0; j2 < 4; ++j2)
#pragma unroll
    for (int r = 0; r < 8; ++r) {
      int row = r + 8 * hi;
      float val = o[j2][r] / lrow[r];
      Obase[(size_t)row * D_MODEL + j2 * 16 + ln] = f32_to_bf16(val);
    }
}

// ---------- launch ----------

extern "C" void kernel_launch(void* const* d_in, const int* in_sizes, int n_in,
                              void* d_out, int out_size, void* d_ws, size_t ws_size,
                              hipStream_t stream) {
  const float* q   = (const float*)d_in[0];
  const float* k   = (const float*)d_in[1];
  const float* v   = (const float*)d_in[2];
  const float* w_q = (const float*)d_in[3];
  const float* b_q = (const float*)d_in[4];
  const float* w_k = (const float*)d_in[5];
  const float* b_k = (const float*)d_in[6];
  const float* w_v = (const float*)d_in[7];
  const float* b_v = (const float*)d_in[8];
  const float* w_o = (const float*)d_in[9];
  const float* b_o = (const float*)d_in[10];
  float* out = (float*)d_out;

  unsigned short* ws = (unsigned short*)d_ws;
  const size_t ACT = (size_t)ROWS * D_MODEL;     // 4M elems
  const size_t WMT = (size_t)D_MODEL * D_MODEL;  // 1M elems
  unsigned short* qb  = ws;
  unsigned short* kb  = qb  + ACT;
  unsigned short* vb  = kb  + ACT;
  unsigned short* wqb = vb  + ACT;
  unsigned short* wkb = wqb + WMT;
  unsigned short* wvb = wkb + WMT;
  unsigned short* wob = wvb + WMT;
  unsigned short* Qp  = wob + WMT;
  unsigned short* Kp  = Qp  + ACT;
  unsigned short* Vp  = Kp  + ACT;
  unsigned short* Ao  = Vp  + ACT;

  cast_f32_to_bf16_kernel<<<dim3((unsigned)(ACT / 256)), 256, 0, stream>>>(q, qb, (int)ACT);
  cast_f32_to_bf16_kernel<<<dim3((unsigned)(ACT / 256)), 256, 0, stream>>>(k, kb, (int)ACT);
  cast_f32_to_bf16_kernel<<<dim3((unsigned)(ACT / 256)), 256, 0, stream>>>(v, vb, (int)ACT);
  cast_f32_to_bf16_kernel<<<dim3((unsigned)(WMT / 256)), 256, 0, stream>>>(w_q, wqb, (int)WMT);
  cast_f32_to_bf16_kernel<<<dim3((unsigned)(WMT / 256)), 256, 0, stream>>>(w_k, wkb, (int)WMT);
  cast_f32_to_bf16_kernel<<<dim3((unsigned)(WMT / 256)), 256, 0, stream>>>(w_v, wvb, (int)WMT);
  cast_f32_to_bf16_kernel<<<dim3((unsigned)(WMT / 256)), 256, 0, stream>>>(w_o, wob, (int)WMT);

  dim3 gg(D_MODEL / BN, ROWS / BM);
  gemm_xwT_bias_kernel<<<gg, 256, 0, stream>>>(qb, wqb, b_q, Qp, nullptr,
                                               ROWS, D_MODEL, D_MODEL);
  gemm_xwT_bias_kernel<<<gg, 256, 0, stream>>>(kb, wkb, b_k, Kp, nullptr,
                                               ROWS, D_MODEL, D_MODEL);
  gemm_xwT_bias_kernel<<<gg, 256, 0, stream>>>(vb, wvb, b_v, Vp, nullptr,
                                               ROWS, D_MODEL, D_MODEL);

  flash_attn_kernel<<<dim3(SEQ / QT, N_HEAD, BATCH), 256, 0, stream>>>(Qp, Kp, Vp, Ao);

  gemm_xwT_bias_kernel<<<gg, 256, 0, stream>>>(Ao, wob, b_o, nullptr, out,
                                               ROWS, D_MODEL, D_MODEL);
}
